// Encoder_84696755077494
// MI455X (gfx1250) — matrix operand order repaired and checked by reading, not compiled, not run on hardware
//
#include <hip/hip_runtime.h>

// GVP-GNN encoder for MI455X (gfx1250, wave32, WMMA).
// - Prep kernel packs each Ws into padded transposed f16 Bt[Npad][Kpad] in d_ws,
//   so WMMA B fragments are single contiguous 32B unconditional global loads.
// - Edge kernel: gather + 3 fused GVPs per edge, masked mean over K -> dh.
// - Node kernel: LN -> GVP(16->32) -> GVP(32->16) -> LN -> mask -> h_V.
// Scalar-channel GEMMs on v_wmma_f32_16x16x32_f16 (f16 in, f32 acc).

#define BB    4
#define NN    1024
#define KK    32
#define DNODE 148   // 16*3 + 100
#define DEDGE 35    // 1*3 + 32
#define EPSF  1e-8f

typedef __attribute__((ext_vector_type(16))) _Float16 v16h;
typedef __attribute__((ext_vector_type(8)))  _Float16 v8h;
typedef __attribute__((ext_vector_type(8)))  float    v8f;

// ---------------- WMMA fragment helpers (A/B: contiguous k order; C/D: ISA layout) ----------------

// A: 16x32 f16 tile in LDS (row-major, leading dim ldK halves).
// lane<16: row=lane, K 0..15; lane>=16: row=lane-16, K 16..31 (matches B).
// Each lane's elements = two contiguous 8-half runs -> 2x ds_load_b128.
__device__ __forceinline__ v16h load_A_frag(const _Float16* As, int ldK, int kbase, int lane) {
  const int r    = lane & 15;
  const int half = (lane & 16) ? 16 : 0;
  const _Float16* p = As + r * ldK + kbase + half;
  v8h lo = *(const v8h*)p;
  v8h hi = *(const v8h*)(p + 8);
  return __builtin_shufflevector(lo, hi, 0, 1, 2, 3, 4, 5, 6, 7,
                                 8, 9, 10, 11, 12, 13, 14, 15);
}

// B: 32x16 from packed transposed f16 weights Bt[Npad][Kpad] in global.
// lane<16: col=lane, K=kbase+0..15; lane>=16: col=lane-16, K=kbase+16..31.
// 16 halves contiguous and 32B-aligned -> one unconditional vector load.
__device__ __forceinline__ v16h load_B_frag(const _Float16* __restrict__ Bt, int Kpad,
                                            int kbase, int nbase, int lane) {
  const int col = lane & 15;
  const int k0  = kbase + ((lane & 16) ? 16 : 0);
  return *(const v16h*)(Bt + (nbase + col) * Kpad + k0);
}

// C/D: 16x16 f32; VGPR i: lanes0-15 -> M=i, lanes16-31 -> M=i+8, col = lane%16.
__device__ __forceinline__ void store_C(float* Cs, int ldN, int nbase, int lane, v8f acc) {
  const int col = lane & 15;
  const int rb  = (lane & 16) ? 8 : 0;
#pragma unroll
  for (int i = 0; i < 8; ++i) Cs[(rb + i) * ldN + nbase + col] = acc[i];
}

// C[16 x ntiles*16] = A[16 x kchunks*32] @ Bt^T, f32 accumulate.
__device__ __forceinline__ void wmma_gemm(const _Float16* As, int ldK, int kchunks,
                                          const _Float16* __restrict__ Bt, int Kpad,
                                          float* Cs, int ldN, int ntiles, int lane) {
  __builtin_prefetch(Bt, 0, 1);  // global_prefetch_b8
  for (int nt = 0; nt < ntiles; ++nt) {
    v8f acc = {};
    for (int kc = 0; kc < kchunks; ++kc) {
      v16h a = load_A_frag(As, ldK, kc * 32, lane);
      v16h b = load_B_frag(Bt, Kpad, kc * 32, nt * 16, lane);
      acc = __builtin_amdgcn_wmma_f32_16x16x32_f16(false, a, false, b, (short)0, acc,
                                                   false, false);
    }
    store_C(Cs, ldN, nt * 16, lane, acc);
  }
}

// ---------------- weight pack: fp32 W[K][N] -> f16 Bt[Npad][Kpad] (zero padded) ----
__global__ void __launch_bounds__(256)
pack_weights(const float* __restrict__ W, int K, int N,
             _Float16* __restrict__ Bt, int Kpad, int Npad) {
  const int total = Npad * Kpad;
  for (int q = blockIdx.x * blockDim.x + threadIdx.x; q < total;
       q += gridDim.x * blockDim.x) {
    const int n = q / Kpad, k = q % Kpad;
    const float v = (n < N && k < K) ? W[k * N + n] : 0.f;
    Bt[q] = (_Float16)v;
  }
}

// ---------------- parameter bundles ----------------

struct EdgeW {
  const float *Wh1, *Wv1, *bs1;         // GVP1: vi=33,h=33
  const float *Wh2, *Wv2, *bs2;         // GVP2: vi=16,h=16
  const float *Wh3, *Wv3, *bs3;         // GVP3: same, no nonlin
  const _Float16 *Bt1, *Bt2, *Bt3;      // packed Ws: [112][288], [112][128], [112][128]
};
struct NodeW {
  const float *Wh0, *Wv0, *bs0;         // dh0: vi=16,h=32,so=400
  const float *Wh1, *Wv1, *bs1;         // dh1: vi=32,h=32,so=100
  const float *beta0, *gamma0, *beta1, *gamma1;
  const _Float16 *Bt0, *Bt1;            // packed Ws: [400][160], [112][448]
};

// ---------------- edge (message) kernel ----------------
// One wave per node: 32 edges = 2 M-tiles of 16 rows. 4 waves/block.
#define E_AS   0        // f16[16*288]   9216 B
#define E_CS   9216     // f32[16*112]   7168 B
#define E_VB   16384    // f32[16*33*3]  6336 B
#define E_VH   22720    // f32[16*33*3]  6336 B
#define E_WM   29056    // f32[16]         64 B
#define E_DS   29120    // f32[148]       592 B
#define E_BLOB 29760
#define E_SW_BYTES 11776          // 2941 small-weight floats, padded
#define EDGE_LDS (E_SW_BYTES + 4 * E_BLOB)

__global__ void __launch_bounds__(128)
gvp_edge_kernel(const float* __restrict__ hv, const float* __restrict__ he,
                const int* __restrict__ eidx, const int* __restrict__ msk,
                EdgeW w, float* __restrict__ dh) {
  extern __shared__ char smem[];
  float* sw = (float*)smem;
  const int tid = threadIdx.x, lane = tid & 31, wv = tid >> 5;

  {  // stage small fp32 weights to LDS
    const float* src[9] = {w.Wh1, w.Wv1, w.bs1, w.Wh2, w.Wv2, w.bs2, w.Wh3, w.Wv3, w.bs3};
    const int off[9] = {0, 1089, 1617, 1717, 1973, 2229, 2329, 2585, 2841};
    const int cnt[9] = {1089, 528, 100, 256, 256, 100, 256, 256, 100};
    for (int a = 0; a < 9; ++a)
      for (int i = tid; i < cnt[a]; i += blockDim.x) sw[off[a] + i] = src[a][i];
  }
  __syncthreads();
  const float *sWh1 = sw, *sWv1 = sw + 1089, *sbs1 = sw + 1617;
  const float *sWh2 = sw + 1717, *sWv2 = sw + 1973, *sbs2 = sw + 2229;
  const float *sWh3 = sw + 2329, *sWv3 = sw + 2585, *sbs3 = sw + 2841;

  char* wb = smem + E_SW_BYTES + wv * E_BLOB;
  _Float16* As = (_Float16*)(wb + E_AS);
  float* Cs = (float*)(wb + E_CS);
  float* Vb = (float*)(wb + E_VB);
  float* Vh = (float*)(wb + E_VH);
  float* wm = (float*)(wb + E_WM);
  float* ds = (float*)(wb + E_DS);

  const int gnode = blockIdx.x * 4 + wv;  // == b*N + n
  const int b = gnode / NN;
  const float mask_n = (float)msk[gnode];

  for (int c = lane; c < DNODE; c += 32) ds[c] = 0.f;

  for (int t = 0; t < 2; ++t) {
    const int ebase = gnode * KK + t * 16;
    // mask_attend per row
    for (int r = lane; r < 16; r += 32)
      wm[r] = mask_n * (float)msk[b * NN + eidx[ebase + r]];
    // ---- A1 = [hV.s(100) | hE.s(32) | hVj.s(100) | vn(33) later | pad 23] ----
    for (int q = lane; q < 16 * 100; q += 32) {
      const int r = q / 100, c = q % 100;
      As[r * 288 + c] = (_Float16)hv[gnode * DNODE + 48 + c];
    }
    for (int q = lane; q < 16 * 32; q += 32) {
      const int r = q / 32, c = q % 32;
      As[r * 288 + 100 + c] = (_Float16)he[(ebase + r) * DEDGE + 3 + c];
    }
    for (int q = lane; q < 16 * 100; q += 32) {
      const int r = q / 100, c = q % 100;
      const int j = eidx[ebase + r];
      As[r * 288 + 132 + c] = (_Float16)hv[(b * NN + j) * DNODE + 48 + c];
    }
    for (int q = lane; q < 16 * 23; q += 32) {
      const int r = q / 23, c = q % 23;
      As[r * 288 + 265 + c] = (_Float16)0.f;
    }
    // ---- V1 = [hV.V(16) | hE.V(1) | hVj.V(16)]  (Vb[r][33][3]) ----
    for (int q = lane; q < 16 * 48; q += 32) {
      const int r = q / 48, i = q % 48;
      Vb[r * 99 + i] = hv[gnode * DNODE + i];
    }
    for (int q = lane; q < 16 * 3; q += 32) {
      const int r = q / 3, c = q % 3;
      Vb[r * 99 + 48 + c] = he[(ebase + r) * DEDGE + c];
    }
    for (int q = lane; q < 16 * 48; q += 32) {
      const int r = q / 48, i = q % 48;
      const int j = eidx[ebase + r];
      Vb[r * 99 + 51 + i] = hv[(b * NN + j) * DNODE + i];
    }
    __syncthreads();
    // ---- GVP1 ----
    for (int q = lane; q < 16 * 99; q += 32) {  // Vh = V @ Wh1 (per coord)
      const int r = q / 99, rem = q % 99, h = rem / 3, c = rem % 3;
      float acc = 0.f;
      for (int v = 0; v < 33; ++v) acc += Vb[r * 99 + v * 3 + c] * sWh1[v * 33 + h];
      Vh[q] = acc;
    }
    __syncthreads();
    for (int q = lane; q < 16 * 33; q += 32) {  // vn -> A cols 232..264
      const int r = q / 33, h = q % 33;
      float s = EPSF;
      for (int c = 0; c < 3; ++c) { float x = Vh[r * 99 + h * 3 + c]; s += x * x; }
      As[r * 288 + 232 + h] = (_Float16)sqrtf(s);
    }
    __syncthreads();
    wmma_gemm(As, 288, 9, w.Bt1, 288, Cs, 112, 7, lane);
    __syncthreads();
    for (int q = lane; q < 16 * 100; q += 32) {  // bias + relu -> s1 (in place)
      const int r = q / 100, c = q % 100;
      Cs[r * 112 + c] = fmaxf(Cs[r * 112 + c] + sbs1[c], 0.f);
    }
    for (int q = lane; q < 16 * 16; q += 32) {   // Vo1 = Vh @ Wv1, sigmoid gate
      const int r = q / 16, v = q % 16;
      float x = 0, y = 0, z = 0;
      for (int h = 0; h < 33; ++h) {
        const float ww = sWv1[h * 16 + v];
        x += Vh[r * 99 + h * 3 + 0] * ww;
        y += Vh[r * 99 + h * 3 + 1] * ww;
        z += Vh[r * 99 + h * 3 + 2] * ww;
      }
      const float nr = sqrtf(x * x + y * y + z * z + EPSF);
      const float g = 1.f / (1.f + __expf(-nr));
      Vb[r * 99 + v * 3 + 0] = g * x;
      Vb[r * 99 + v * 3 + 1] = g * y;
      Vb[r * 99 + v * 3 + 2] = g * z;
    }
    __syncthreads();
    // ---- GVP2 ----
    for (int q = lane; q < 16 * 100; q += 32) {
      const int r = q / 100, c = q % 100;
      As[r * 288 + c] = (_Float16)Cs[r * 112 + c];
    }
    for (int q = lane; q < 16 * 12; q += 32) {
      const int r = q / 12, c = q % 12;
      As[r * 288 + 116 + c] = (_Float16)0.f;  // 100..115 = vn2 later
    }
    for (int q = lane; q < 16 * 48; q += 32) {
      const int r = q / 48, rem = q % 48, h = rem / 3, c = rem % 3;
      float acc = 0.f;
      for (int v = 0; v < 16; ++v) acc += Vb[r * 99 + v * 3 + c] * sWh2[v * 16 + h];
      Vh[r * 99 + h * 3 + c] = acc;
    }
    __syncthreads();
    for (int q = lane; q < 16 * 16; q += 32) {
      const int r = q / 16, h = q % 16;
      float s = EPSF;
      for (int c = 0; c < 3; ++c) { float x = Vh[r * 99 + h * 3 + c]; s += x * x; }
      As[r * 288 + 100 + h] = (_Float16)sqrtf(s);
    }
    __syncthreads();
    wmma_gemm(As, 288, 4, w.Bt2, 128, Cs, 112, 7, lane);
    __syncthreads();
    for (int q = lane; q < 16 * 100; q += 32) {
      const int r = q / 100, c = q % 100;
      Cs[r * 112 + c] = fmaxf(Cs[r * 112 + c] + sbs2[c], 0.f);
    }
    for (int q = lane; q < 16 * 16; q += 32) {
      const int r = q / 16, v = q % 16;
      float x = 0, y = 0, z = 0;
      for (int h = 0; h < 16; ++h) {
        const float ww = sWv2[h * 16 + v];
        x += Vh[r * 99 + h * 3 + 0] * ww;
        y += Vh[r * 99 + h * 3 + 1] * ww;
        z += Vh[r * 99 + h * 3 + 2] * ww;
      }
      const float nr = sqrtf(x * x + y * y + z * z + EPSF);
      const float g = 1.f / (1.f + __expf(-nr));
      Vb[r * 99 + v * 3 + 0] = g * x;
      Vb[r * 99 + v * 3 + 1] = g * y;
      Vb[r * 99 + v * 3 + 2] = g * z;
    }
    __syncthreads();
    // ---- GVP3 (no relu, no gate) ----
    for (int q = lane; q < 16 * 100; q += 32) {
      const int r = q / 100, c = q % 100;
      As[r * 288 + c] = (_Float16)Cs[r * 112 + c];
    }
    for (int q = lane; q < 16 * 12; q += 32) {
      const int r = q / 12, c = q % 12;
      As[r * 288 + 116 + c] = (_Float16)0.f;
    }
    for (int q = lane; q < 16 * 48; q += 32) {
      const int r = q / 48, rem = q % 48, h = rem / 3, c = rem % 3;
      float acc = 0.f;
      for (int v = 0; v < 16; ++v) acc += Vb[r * 99 + v * 3 + c] * sWh3[v * 16 + h];
      Vh[r * 99 + h * 3 + c] = acc;
    }
    __syncthreads();
    for (int q = lane; q < 16 * 16; q += 32) {
      const int r = q / 16, h = q % 16;
      float s = EPSF;
      for (int c = 0; c < 3; ++c) { float x = Vh[r * 99 + h * 3 + c]; s += x * x; }
      As[r * 288 + 100 + h] = (_Float16)sqrtf(s);
    }
    __syncthreads();
    wmma_gemm(As, 288, 4, w.Bt3, 128, Cs, 112, 7, lane);
    __syncthreads();
    for (int q = lane; q < 16 * 100; q += 32) {
      const int r = q / 100, c = q % 100;
      Cs[r * 112 + c] += sbs3[c];
    }
    for (int q = lane; q < 16 * 16; q += 32) {
      const int r = q / 16, v = q % 16;
      float x = 0, y = 0, z = 0;
      for (int h = 0; h < 16; ++h) {
        const float ww = sWv3[h * 16 + v];
        x += Vh[r * 99 + h * 3 + 0] * ww;
        y += Vh[r * 99 + h * 3 + 1] * ww;
        z += Vh[r * 99 + h * 3 + 2] * ww;
      }
      Vb[r * 99 + v * 3 + 0] = x;
      Vb[r * 99 + v * 3 + 1] = y;
      Vb[r * 99 + v * 3 + 2] = z;
    }
    __syncthreads();
    // masked sum over this tile's 16 edges
    for (int c = lane; c < DNODE; c += 32) {
      float s = 0.f;
      for (int r = 0; r < 16; ++r) {
        const float val = (c < 48) ? Vb[r * 99 + c] : Cs[r * 112 + (c - 48)];
        s += wm[r] * val;
      }
      ds[c] += s;
    }
    __syncthreads();
  }
  for (int c = lane; c < DNODE; c += 32)
    dh[gnode * DNODE + c] = ds[c] * (1.f / (float)KK);
}

// ---------------- node (update) kernel ----------------
// One wave per 16-node M-tile; 4 waves/block.
#define N_AS   0        // f16[16*448]  14336 B
#define N_CS   14336    // f32[16*400]  25600 B
#define N_VB   39936    // f32[16*32*3]  6144 B
#define N_VH   46080    // f32[16*32*3]  6144 B
#define N_ST   52224    // f32[48]        192 B
#define N_BLOB 52480
#define N_SW_BYTES 15904         // 3972 floats padded
#define NODE_LDS (N_SW_BYTES + 4 * N_BLOB)

__global__ void __launch_bounds__(128)
gvp_node_kernel(const float* __restrict__ dh, const int* __restrict__ msk,
                NodeW w, float* __restrict__ out) {
  extern __shared__ char smem[];
  float* sw = (float*)smem;
  const int tid = threadIdx.x, lane = tid & 31, wv = tid >> 5;

  {
    const float* src[10] = {w.Wh0, w.Wv0, w.bs0, w.Wh1, w.Wv1, w.bs1,
                            w.beta0, w.gamma0, w.beta1, w.gamma1};
    const int off[10] = {0, 512, 1536, 1936, 2960, 3472, 3572, 3672, 3772, 3872};
    const int cnt[10] = {512, 1024, 400, 1024, 512, 100, 100, 100, 100, 100};
    for (int a = 0; a < 10; ++a)
      for (int i = tid; i < cnt[a]; i += blockDim.x) sw[off[a] + i] = src[a][i];
  }
  __syncthreads();
  const float *sWh0 = sw, *sWv0 = sw + 512, *sbs0 = sw + 1536;
  const float *sWh1 = sw + 1936, *sWv1 = sw + 2960, *sbs1 = sw + 3472;
  const float *sbe0 = sw + 3572, *sga0 = sw + 3672, *sbe1 = sw + 3772, *sga1 = sw + 3872;

  char* wb = smem + N_SW_BYTES + wv * N_BLOB;
  _Float16* As = (_Float16*)(wb + N_AS);
  float* Cs = (float*)(wb + N_CS);
  float* Vb = (float*)(wb + N_VB);
  float* Vh = (float*)(wb + N_VH);
  float* st = (float*)(wb + N_ST);

  const int base = (blockIdx.x * 4 + wv) * 16;

  // layernorm1 stats (one row per lane<16)
  if (lane < 16) {
    const float* x = dh + (base + lane) * DNODE;
    float vs = 0.f;
    for (int i = 0; i < 48; ++i) vs += x[i] * x[i];
    float mu = 0.f;
    for (int i = 0; i < 100; ++i) mu += x[48 + i];
    mu *= 0.01f;
    float var = 0.f;
    for (int i = 0; i < 100; ++i) { float d = x[48 + i] - mu; var += d * d; }
    var *= 0.01f;
    st[lane * 3 + 0] = 1.f / sqrtf(vs / 16.f + EPSF);
    st[lane * 3 + 1] = mu;
    st[lane * 3 + 2] = 1.f / sqrtf(var + 1e-5f);
  }
  __syncthreads();
  for (int q = lane; q < 16 * 48; q += 32) {  // normed V (v<16) into Vb [r][32][3]
    const int r = q / 48, i = q % 48;
    Vb[r * 96 + i] = dh[(base + r) * DNODE + i] * st[r * 3];
  }
  for (int q = lane; q < 16 * 100; q += 32) {  // A = [LN(s)*g+b | vn(32) later | pad 28]
    const int r = q / 100, c = q % 100;
    const float v = (dh[(base + r) * DNODE + 48 + c] - st[r * 3 + 1]) * st[r * 3 + 2]
                    * sga0[c] + sbe0[c];
    As[r * 448 + c] = (_Float16)v;
  }
  for (int q = lane; q < 16 * 28; q += 32) {
    const int r = q / 28, c = q % 28;
    As[r * 448 + 132 + c] = (_Float16)0.f;
  }
  __syncthreads();
  // ---- GVP dh0 (vi=16 -> h=32, so=400, relu+gate) ----
  for (int q = lane; q < 16 * 96; q += 32) {
    const int r = q / 96, rem = q % 96, h = rem / 3, c = rem % 3;
    float acc = 0.f;
    for (int v = 0; v < 16; ++v) acc += Vb[(r * 32 + v) * 3 + c] * sWh0[v * 32 + h];
    Vh[(r * 32 + h) * 3 + c] = acc;
  }
  __syncthreads();
  for (int q = lane; q < 16 * 32; q += 32) {
    const int r = q / 32, h = q % 32;
    float s = EPSF;
    for (int c = 0; c < 3; ++c) { float x = Vh[(r * 32 + h) * 3 + c]; s += x * x; }
    As[r * 448 + 100 + h] = (_Float16)sqrtf(s);
  }
  __syncthreads();
  wmma_gemm(As, 448, 5, w.Bt0, 160, Cs, 400, 25, lane);
  __syncthreads();
  for (int q = lane; q < 16 * 400; q += 32)
    Cs[q] = fmaxf(Cs[q] + sbs0[q % 400], 0.f);
  for (int q = lane; q < 16 * 32; q += 32) {
    const int r = q / 32, v = q % 32;
    float x = 0, y = 0, z = 0;
    for (int h = 0; h < 32; ++h) {
      const float ww = sWv0[h * 32 + v];
      x += Vh[(r * 32 + h) * 3 + 0] * ww;
      y += Vh[(r * 32 + h) * 3 + 1] * ww;
      z += Vh[(r * 32 + h) * 3 + 2] * ww;
    }
    const float nr = sqrtf(x * x + y * y + z * z + EPSF);
    const float g = 1.f / (1.f + __expf(-nr));
    Vb[(r * 32 + v) * 3 + 0] = g * x;
    Vb[(r * 32 + v) * 3 + 1] = g * y;
    Vb[(r * 32 + v) * 3 + 2] = g * z;
  }
  __syncthreads();
  // ---- GVP dh1 (vi=32 -> h=32, so=100, no nonlin) ----
  for (int q = lane; q < 16 * 400; q += 32) {
    const int r = q / 400, c = q % 400;
    As[r * 448 + c] = (_Float16)Cs[r * 400 + c];
  }
  for (int q = lane; q < 16 * 16; q += 32) {
    const int r = q / 16, c = q % 16;
    As[r * 448 + 432 + c] = (_Float16)0.f;  // 400..431 = vn later
  }
  for (int q = lane; q < 16 * 96; q += 32) {
    const int r = q / 96, rem = q % 96, h = rem / 3, c = rem % 3;
    float acc = 0.f;
    for (int v = 0; v < 32; ++v) acc += Vb[(r * 32 + v) * 3 + c] * sWh1[v * 32 + h];
    Vh[(r * 32 + h) * 3 + c] = acc;
  }
  __syncthreads();
  for (int q = lane; q < 16 * 32; q += 32) {
    const int r = q / 32, h = q % 32;
    float s = EPSF;
    for (int c = 0; c < 3; ++c) { float x = Vh[(r * 32 + h) * 3 + c]; s += x * x; }
    As[r * 448 + 400 + h] = (_Float16)sqrtf(s);
  }
  __syncthreads();
  wmma_gemm(As, 448, 14, w.Bt1, 448, Cs, 112, 7, lane);
  __syncthreads();
  for (int q = lane; q < 16 * 100; q += 32) {
    const int r = q / 100, c = q % 100;
    Cs[r * 112 + c] += sbs1[c];
  }
  for (int q = lane; q < 16 * 16; q += 32) {
    const int r = q / 16, v = q % 16;
    float x = 0, y = 0, z = 0;
    for (int h = 0; h < 32; ++h) {
      const float ww = sWv1[h * 16 + v];
      x += Vh[(r * 32 + h) * 3 + 0] * ww;
      y += Vh[(r * 32 + h) * 3 + 1] * ww;
      z += Vh[(r * 32 + h) * 3 + 2] * ww;
    }
    Vb[(r * 32 + v) * 3 + 0] = x;
    Vb[(r * 32 + v) * 3 + 1] = y;
    Vb[(r * 32 + v) * 3 + 2] = z;
  }
  __syncthreads();
  // layernorm2 stats
  if (lane < 16) {
    const int r = lane;
    float vs = 0.f;
    for (int i = 0; i < 48; ++i) { float x = Vb[r * 96 + i]; vs += x * x; }
    float mu = 0.f;
    for (int i = 0; i < 100; ++i) mu += Cs[r * 112 + i];
    mu *= 0.01f;
    float var = 0.f;
    for (int i = 0; i < 100; ++i) { float d = Cs[r * 112 + i] - mu; var += d * d; }
    var *= 0.01f;
    st[r * 3 + 0] = 1.f / sqrtf(vs / 16.f + EPSF);
    st[r * 3 + 1] = mu;
    st[r * 3 + 2] = 1.f / sqrtf(var + 1e-5f);
  }
  __syncthreads();
  for (int q = lane; q < 16 * DNODE; q += 32) {
    const int r = q / DNODE, c = q % DNODE;
    const int gn = base + r;
    const float mf = (float)msk[gn];
    float v;
    if (c < 48) v = Vb[r * 96 + c] * st[r * 3];
    else        v = (Cs[r * 112 + (c - 48)] - st[r * 3 + 1]) * st[r * 3 + 2]
                    * sga1[c - 48] + sbe1[c - 48];
    out[gn * DNODE + c] = mf * v;
  }
}

// ---------------- launch ----------------

// packed-weight layout in d_ws (halves), after dh scratch
#define DH_BYTES   (BB * NN * DNODE * 4)          // 2,424,832 (64B aligned)
#define PW_E1_OFF  0                              // [112][288] = 32256 halves
#define PW_E2_OFF  32256                          // [112][128] = 14336
#define PW_E3_OFF  46592                          // [112][128] = 14336
#define PW_N0_OFF  60928                          // [400][160] = 64000
#define PW_N1_OFF  124928                         // [112][448] = 50176
#define PW_LAYER   175104                         // halves per layer

extern "C" void kernel_launch(void* const* d_in, const int* in_sizes, int n_in,
                              void* d_out, int out_size, void* d_ws, size_t ws_size,
                              hipStream_t stream) {
  (void)in_sizes; (void)n_in; (void)out_size; (void)ws_size;
  const float* hV  = (const float*)d_in[0];
  const float* hE  = (const float*)d_in[1];
  const int*   eid = (const int*)d_in[2];
  const int*   msk = (const int*)d_in[3];
  float* out = (float*)d_out;
  float* dh  = (float*)d_ws;
  _Float16* pw = (_Float16*)((char*)d_ws + DH_BYTES);

  hipFuncSetAttribute((const void*)gvp_edge_kernel,
                      hipFuncAttributeMaxDynamicSharedMemorySize, EDGE_LDS);
  hipFuncSetAttribute((const void*)gvp_node_kernel,
                      hipFuncAttributeMaxDynamicSharedMemorySize, NODE_LDS);

  // pack all Ws matrices (fp32 row-major -> padded transposed f16)
  for (int l = 0; l < 3; ++l) {
    void* const* P = d_in + 4 + l * 24;
    _Float16* pl = pw + l * PW_LAYER;
    pack_weights<<<126, 256, 0, stream>>>((const float*)P[1], 265, 100,
                                          pl + PW_E1_OFF, 288, 112);
    pack_weights<<<56, 256, 0, stream>>>((const float*)P[5], 116, 100,
                                         pl + PW_E2_OFF, 128, 112);
    pack_weights<<<56, 256, 0, stream>>>((const float*)P[9], 116, 100,
                                         pl + PW_E3_OFF, 128, 112);
    pack_weights<<<250, 256, 0, stream>>>((const float*)P[13], 132, 400,
                                          pl + PW_N0_OFF, 160, 400);
    pack_weights<<<196, 256, 0, stream>>>((const float*)P[17], 432, 100,
                                          pl + PW_N1_OFF, 448, 112);
  }

  for (int l = 0; l < 3; ++l) {
    void* const* P = d_in + 4 + l * 24;  // pytree order: W_EV[0..2]{Wh,Ws,Wv,bs},
                                         // W_dh[0..1]{...}, norm[0..1]{beta,gamma}
    _Float16* pl = pw + l * PW_LAYER;
    EdgeW ew = {(const float*)P[0], (const float*)P[2],  (const float*)P[3],
                (const float*)P[4], (const float*)P[6],  (const float*)P[7],
                (const float*)P[8], (const float*)P[10], (const float*)P[11],
                pl + PW_E1_OFF, pl + PW_E2_OFF, pl + PW_E3_OFF};
    NodeW nw = {(const float*)P[12], (const float*)P[14], (const float*)P[15],
                (const float*)P[16], (const float*)P[18], (const float*)P[19],
                (const float*)P[20], (const float*)P[21], (const float*)P[22],
                (const float*)P[23], pl + PW_N0_OFF, pl + PW_N1_OFF};
    const float* hv_in = (l == 0) ? hV : out;  // h_V ping-pongs through d_out
    gvp_edge_kernel<<<(BB * NN) / 4, 128, EDGE_LDS, stream>>>(hv_in, hE, eid, msk, ew, dh);
    gvp_node_kernel<<<(BB * NN) / 64, 128, NODE_LDS, stream>>>(dh, msk, nw, out);
  }
}
